// DecoderBlock_37830071943378
// MI455X (gfx1250) — matrix operand order repaired and checked by reading, not compiled, not run on hardware
//
#include <hip/hip_runtime.h>

// ---------------------------------------------------------------------------
// DETR-style decoder block on gfx1250 (MI455X).
// All GEMMs: v_wmma_f32_16x16x32_f16, f16 operands / f32 accumulate.
// B operands are kept/made N-major so every WMMA fragment is a contiguous
// per-lane run -> direct global_load_b128, no LDS staging, no syncs.
// Workspace requirement: ~99.1M floats (~397 MB).
// ---------------------------------------------------------------------------

typedef __attribute__((ext_vector_type(16))) _Float16 v16h;
typedef __attribute__((ext_vector_type(8)))  float    v8f;
typedef __attribute__((ext_vector_type(4)))  float    v4f;

// Pack 4x float4 -> v16h (element j of each v4f -> f16)
__device__ __forceinline__ v16h cvt16(v4f f0, v4f f1, v4f f2, v4f f3) {
    v16h r;
    #pragma unroll
    for (int j = 0; j < 4; ++j) {
        r[j]      = (_Float16)f0[j];
        r[4 + j]  = (_Float16)f1[j];
        r[8 + j]  = (_Float16)f2[j];
        r[12 + j] = (_Float16)f3[j];
    }
    return r;
}

// A fragment (16x32 f16), lane l -> row = l&15, sub = l>>4.
// elem j : K = sub*16 + j. p = row base + k0.
__device__ __forceinline__ v16h frag_a_full(const float* p, int sub) {
    const float* q = p + sub * 16;
    return cvt16(*(const v4f*)(q),      *(const v4f*)(q + 4),
                 *(const v4f*)(q + 8), *(const v4f*)(q + 12));
}
// B fragment (32x16 f16) from N-major Bt, lane l -> N-row = l&15.
// elem j : K = sub*16 + j  (16 contiguous floats).
__device__ __forceinline__ v16h frag_b_full(const float* p, int sub) {
    const float* q = p + sub * 16;
    return cvt16(*(const v4f*)(q),     *(const v4f*)(q + 4),
                 *(const v4f*)(q + 8), *(const v4f*)(q + 12));
}
// K-tail variants: clamp+select (branchless), kr = remaining K in [1,31].
__device__ __forceinline__ v16h frag_a_tail(const float* p, int sub, int kr) {
    v16h r;
    #pragma unroll
    for (int j = 0; j < 16; ++j) {
        int koff = sub * 16 + j;
        int kc = koff < kr ? koff : 0;
        float x = p[kc];
        r[j] = (_Float16)((koff < kr) ? x : 0.f);
    }
    return r;
}
__device__ __forceinline__ v16h frag_b_tail(const float* p, int sub, int kr) {
    v16h r;
    #pragma unroll
    for (int j = 0; j < 16; ++j) {
        int koff = sub * 16 + j;
        int kc = koff < kr ? koff : 0;
        float x = p[kc];
        r[j] = (_Float16)((koff < kr) ? x : 0.f);
    }
    return r;
}

__device__ __forceinline__ void store_tile(float* __restrict__ C, v8f acc,
                                           int mt, int nt, int M, int N, int ldc,
                                           const float* __restrict__ bias,
                                           float alpha, int relu, int sub, int lo)
{
    int gc = nt + lo;
    if (gc < N) {
        float bval = bias ? bias[gc] : 0.f;
        #pragma unroll
        for (int r = 0; r < 8; ++r) {
            int gr = mt + r + 8 * sub;
            if (gr < M) {
                float v = alpha * acc[r] + bval;
                if (relu) v = fmaxf(v, 0.f);
                C[(long long)gr * ldc + gc] = v;
            }
        }
    }
}

// ---------------------------------------------------------------------------
// C = alpha * (A @ Bt^T) + bias, optional ReLU. A: MxK row-major (lda),
// Bt: NxK row-major (ldb), C: MxN (ldc). One wave per 32x32 C tile: 2x2
// register blocking, 4 WMMAs per 32-K chunk, fragments loaded straight from
// global as b128. Row indices clamped (OOB rows feed unstored C entries);
// only the K-tail masks. Dual batch strides address (b,h) head views.
// Requires lda, ldb multiples of 4 (all pitches padded accordingly).
// ---------------------------------------------------------------------------
__global__ void __launch_bounds__(32)
wmma_gemm_kernel(const float* __restrict__ A, const float* __restrict__ Bt,
                 const float* __restrict__ bias, float* __restrict__ C,
                 int M, int N, int K, int lda, int ldb, int ldc,
                 long long sa_o, long long sa_i, long long sb_o, long long sb_i,
                 long long sc_o, long long sc_i,
                 int inner, float alpha, int relu)
{
    int z  = blockIdx.z;
    int zo = z / inner, zi = z - zo * inner;
    A  += zo * sa_o + zi * sa_i;
    Bt += zo * sb_o + zi * sb_i;
    C  += zo * sc_o + zi * sc_i;

    int n0 = blockIdx.x * 32, m0 = blockIdx.y * 32;
    int l = threadIdx.x, sub = l >> 4, lo = l & 15;

    int ar0 = m0 + lo;      ar0 = ar0 < M - 1 ? ar0 : M - 1;
    int ar1 = m0 + 16 + lo; ar1 = ar1 < M - 1 ? ar1 : M - 1;
    int br0 = n0 + lo;      br0 = br0 < N - 1 ? br0 : N - 1;
    int br1 = n0 + 16 + lo; br1 = br1 < N - 1 ? br1 : N - 1;
    const float* pa0 = A + (long long)ar0 * lda;
    const float* pa1 = A + (long long)ar1 * lda;
    const float* pb0 = Bt + (long long)br0 * ldb;
    const float* pb1 = Bt + (long long)br1 * ldb;

    v8f c00 = {0.f,0.f,0.f,0.f,0.f,0.f,0.f,0.f};
    v8f c01 = c00, c10 = c00, c11 = c00;

    int kfull = K & ~31;
    for (int k0 = 0; k0 < kfull; k0 += 32) {
        __builtin_prefetch(pa0 + k0 + 32, 0, 1);
        v16h a0 = frag_a_full(pa0 + k0, sub);
        v16h a1 = frag_a_full(pa1 + k0, sub);
        v16h b0 = frag_b_full(pb0 + k0, sub);
        v16h b1 = frag_b_full(pb1 + k0, sub);
        c00 = __builtin_amdgcn_wmma_f32_16x16x32_f16(false, a0, false, b0, (short)0, c00, false, false);
        c01 = __builtin_amdgcn_wmma_f32_16x16x32_f16(false, a0, false, b1, (short)0, c01, false, false);
        c10 = __builtin_amdgcn_wmma_f32_16x16x32_f16(false, a1, false, b0, (short)0, c10, false, false);
        c11 = __builtin_amdgcn_wmma_f32_16x16x32_f16(false, a1, false, b1, (short)0, c11, false, false);
    }
    if (kfull < K) {
        int kr = K - kfull;
        v16h a0 = frag_a_tail(pa0 + kfull, sub, kr);
        v16h a1 = frag_a_tail(pa1 + kfull, sub, kr);
        v16h b0 = frag_b_tail(pb0 + kfull, sub, kr);
        v16h b1 = frag_b_tail(pb1 + kfull, sub, kr);
        c00 = __builtin_amdgcn_wmma_f32_16x16x32_f16(false, a0, false, b0, (short)0, c00, false, false);
        c01 = __builtin_amdgcn_wmma_f32_16x16x32_f16(false, a0, false, b1, (short)0, c01, false, false);
        c10 = __builtin_amdgcn_wmma_f32_16x16x32_f16(false, a1, false, b0, (short)0, c10, false, false);
        c11 = __builtin_amdgcn_wmma_f32_16x16x32_f16(false, a1, false, b1, (short)0, c11, false, false);
    }

    store_tile(C, c00, m0,      n0,      M, N, ldc, bias, alpha, relu, sub, lo);
    store_tile(C, c01, m0,      n0 + 16, M, N, ldc, bias, alpha, relu, sub, lo);
    store_tile(C, c10, m0 + 16, n0,      M, N, ldc, bias, alpha, relu, sub, lo);
    store_tile(C, c11, m0 + 16, n0 + 16, M, N, ldc, bias, alpha, relu, sub, lo);
}

// ---------------------------------------------------------------------------
// Transposes (one-shot, bandwidth-trivial).
// ---------------------------------------------------------------------------
// Wt[c*R + r] = W[r*Cc + c]   (weight K x N -> N x K)
__global__ void transpose_kernel(const float* __restrict__ in, float* __restrict__ out,
                                 int R, int Cc)
{
    long long i = (long long)blockIdx.x * blockDim.x + threadIdx.x;
    long long total = (long long)R * Cc;
    if (i >= total) return;
    long long c = i / R, r = i - c * R;
    out[i] = in[r * Cc + c];
}
// vt[((b*H+h)*64 + d)*S + s] = v[(b*S+s)*D + h*64 + d]
__global__ void transpose_head_kernel(const float* __restrict__ v, float* __restrict__ vt,
                                      int Bn, int Hn, int Sn, int Dn)
{
    long long i = (long long)blockIdx.x * blockDim.x + threadIdx.x;
    long long total = (long long)Bn * Hn * 64 * Sn;
    if (i >= total) return;
    long long s = i % Sn;  long long t1 = i / Sn;
    long long d = t1 % 64; long long t2 = t1 / 64;
    long long h = t2 % Hn; long long b = t2 / Hn;
    vt[i] = v[(b * Sn + s) * Dn + h * 64 + d];
}
// xt[(b*Cc + c)*pitch + r] = x[(b*R + r)*Cc + c]; zero-pad r >= R
__global__ void transpose_b_kernel(const float* __restrict__ x, float* __restrict__ xt,
                                   int Bn, int R, int Cc, int pitch)
{
    long long i = (long long)blockIdx.x * blockDim.x + threadIdx.x;
    long long total = (long long)Bn * Cc * pitch;
    if (i >= total) return;
    long long r = i % pitch; long long t = i / pitch;
    long long c = t % Cc;    long long b = t / Cc;
    xt[i] = (r < R) ? x[(b * R + r) * Cc + c] : 0.f;
}

// ---------------------------------------------------------------------------
// Block reduction helpers (blockDim.x power of two).
// ---------------------------------------------------------------------------
__device__ __forceinline__ float blk_max(float v, float* red) {
    int t = threadIdx.x;
    red[t] = v; __syncthreads();
    for (int s = blockDim.x >> 1; s > 0; s >>= 1) {
        if (t < s) red[t] = fmaxf(red[t], red[t + s]);
        __syncthreads();
    }
    float r = red[0]; __syncthreads();
    return r;
}
__device__ __forceinline__ float blk_sum(float v, float* red) {
    int t = threadIdx.x;
    red[t] = v; __syncthreads();
    for (int s = blockDim.x >> 1; s > 0; s >>= 1) {
        if (t < s) red[t] += red[t + s];
        __syncthreads();
    }
    float r = red[0]; __syncthreads();
    return r;
}

// Dual softmax: probs1 = softmax(s) in place; probs2 = softmax(s - dist_ij).
__global__ void __launch_bounds__(128)
softmax_pair_kernel(float* __restrict__ scores, float* __restrict__ probs2,
                    const float* __restrict__ coords, int Hn, int Sn)
{
    __shared__ float red[128];
    int i = blockIdx.x, h = blockIdx.y, b = blockIdx.z;
    long long rowbase = (((long long)b * Hn + h) * Sn + i) * (long long)Sn;
    const float* cb = coords + (long long)b * Sn * 2;
    float xi = cb[2 * i], yi = cb[2 * i + 1];
    int t = threadIdx.x;

    float v1[4], v2[4];
    float m1 = -3.4e38f, m2 = -3.4e38f;
    int cnt = 0;
    for (int j = t; j < Sn; j += 128) {
        float s  = scores[rowbase + j];
        float dx = xi - cb[2 * j], dy = yi - cb[2 * j + 1];
        float s2 = s - sqrtf(dx * dx + dy * dy);
        v1[cnt] = s; v2[cnt] = s2; ++cnt;
        m1 = fmaxf(m1, s); m2 = fmaxf(m2, s2);
    }
    m1 = blk_max(m1, red);
    m2 = blk_max(m2, red);
    float s1 = 0.f, s2s = 0.f;
    for (int c = 0; c < cnt; ++c) {
        v1[c] = __expf(v1[c] - m1); s1  += v1[c];
        v2[c] = __expf(v2[c] - m2); s2s += v2[c];
    }
    s1  = blk_sum(s1, red);
    s2s = blk_sum(s2s, red);
    float r1 = 1.f / s1, r2 = 1.f / s2s;
    cnt = 0;
    for (int j = t; j < Sn; j += 128) {
        scores[rowbase + j] = v1[cnt] * r1;
        probs2[rowbase + j] = v2[cnt] * r2;
        ++cnt;
    }
}

// Plain softmax, rows of length len (<=64) with pitch ld. Pre-scaled input.
__global__ void __launch_bounds__(64)
softmax_row_kernel(float* __restrict__ x, int len, int ld)
{
    __shared__ float red[64];
    long long row = blockIdx.x;
    float* p = x + row * (long long)ld;
    int t = threadIdx.x;
    float v = (t < len) ? p[t] : -3.4e38f;
    red[t] = v; __syncthreads();
    for (int s = 32; s > 0; s >>= 1) { if (t < s) red[t] = fmaxf(red[t], red[t + s]); __syncthreads(); }
    float mx = red[0]; __syncthreads();
    float e = (t < len) ? __expf(v - mx) : 0.f;
    red[t] = e; __syncthreads();
    for (int s = 32; s > 0; s >>= 1) { if (t < s) red[t] += red[t + s]; __syncthreads(); }
    float sum = red[0];
    if (t < len) p[t] = e / sum;
}

// o = 0.5*LN(oq+o1; g1,b1) + 0.5*LN(oq+o2; g2,b2), row length Dn (=512).
__global__ void __launch_bounds__(128)
fuse_ln_kernel(const float* __restrict__ oq, const float* __restrict__ o1,
               const float* __restrict__ o2,
               const float* __restrict__ g1, const float* __restrict__ b1,
               const float* __restrict__ g2, const float* __restrict__ b2,
               float* __restrict__ o, int Dn)
{
    __shared__ float red[128];
    long long row = blockIdx.x;
    const float* p0 = oq + row * Dn;
    const float* p1 = o1 + row * Dn;
    const float* p2 = o2 + row * Dn;
    int t = threadIdx.x;
    float a1[4], a2[4];
    int cnt = 0;
    float s1 = 0.f, q1 = 0.f, s2 = 0.f, q2 = 0.f;
    for (int j = t; j < Dn; j += 128) {
        float x  = p0[j];
        float y1 = x + p1[j], y2 = x + p2[j];
        a1[cnt] = y1; a2[cnt] = y2; ++cnt;
        s1 += y1; q1 += y1 * y1; s2 += y2; q2 += y2 * y2;
    }
    s1 = blk_sum(s1, red); q1 = blk_sum(q1, red);
    s2 = blk_sum(s2, red); q2 = blk_sum(q2, red);
    float inv = 1.f / (float)Dn;
    float m1 = s1 * inv, m2 = s2 * inv;
    float r1 = rsqrtf(q1 * inv - m1 * m1 + 1e-3f);
    float r2 = rsqrtf(q2 * inv - m2 * m2 + 1e-3f);
    float* po = o + row * Dn;
    cnt = 0;
    for (int j = t; j < Dn; j += 128) {
        float u1 = (a1[cnt] - m1) * r1 * g1[j] + b1[j];
        float u2 = (a2[cnt] - m2) * r2 * g2[j] + b2[j];
        po[j] = 0.5f * u1 + 0.5f * u2;
        ++cnt;
    }
}

// y = LN(x1 + x2; g,bt), row length Dn (=256); strided row pitches.
__global__ void __launch_bounds__(128)
ln_res_kernel(const float* __restrict__ x1, int ld1,
              const float* __restrict__ x2, int ld2,
              const float* __restrict__ g, const float* __restrict__ bt,
              float* __restrict__ y, int ldy, int Dn)
{
    __shared__ float red[128];
    long long row = blockIdx.x;
    const float* p1 = x1 + row * ld1;
    const float* p2 = x2 + row * ld2;
    float* py = y + row * ldy;
    int t = threadIdx.x;
    float a[4]; int cnt = 0; float s = 0.f, qq = 0.f;
    for (int j = t; j < Dn; j += 128) {
        float v = p1[j] + p2[j];
        a[cnt++] = v; s += v; qq += v * v;
    }
    s = blk_sum(s, red); qq = blk_sum(qq, red);
    float inv = 1.f / (float)Dn, m = s * inv;
    float r = rsqrtf(qq * inv - m * m + 1e-3f);
    cnt = 0;
    for (int j = t; j < Dn; j += 128)
        py[j] = (a[cnt++] - m) * r * g[j] + bt[j];
}

// q[row, c] += qp[row, c % 256]
__global__ void add_dup_kernel(float* __restrict__ q, const float* __restrict__ qp,
                               long long total)
{
    long long i = (long long)blockIdx.x * blockDim.x + threadIdx.x;
    if (i >= total) return;
    long long row = i >> 9;
    int c = (int)(i & 511);
    q[i] += qp[row * 256 + (c & 255)];
}

// y[row, h*64+j] = j<32 ? xa[row*lda + h*32+j] : xb[(row%bmod)*256 + h*32+j-32]
__global__ void concat_heads_kernel(const float* __restrict__ xa, int lda,
                                    const float* __restrict__ xb, int bmod,
                                    float* __restrict__ y, long long total)
{
    long long i = (long long)blockIdx.x * blockDim.x + threadIdx.x;
    if (i >= total) return;
    long long row = i >> 9;
    int c = (int)(i & 511);
    int h = c >> 6, j = c & 63;
    long long rb = bmod ? (row % bmod) : row;
    y[i] = (j < 32) ? xa[row * (long long)lda + h * 32 + j]
                    : xb[rb * 256 + h * 32 + (j - 32)];
}

// Sinusoidal embedding (L, Dn): e[n,2i]=sin(n/10000^(2i/Dn)), e[n,2i+1]=cos.
__global__ void sine_embed_kernel(float* __restrict__ e, int L, int Dn)
{
    int idx = blockIdx.x * blockDim.x + threadIdx.x;
    int half = Dn >> 1;
    if (idx >= L * half) return;
    int n = idx / half, i = idx - n * half;
    float div = __powf(10000.f, 2.f * (float)i / (float)Dn);
    float ang = (float)n / div;
    e[n * Dn + 2 * i]     = __sinf(ang);
    e[n * Dn + 2 * i + 1] = __cosf(ang);
}

// ---------------------------------------------------------------------------
static inline void gemm(hipStream_t st, const float* A, const float* Bt,
                        const float* bias, float* C,
                        int M, int N, int K, int lda, int ldb, int ldc,
                        long long sa_o, long long sa_i,
                        long long sb_o, long long sb_i,
                        long long sc_o, long long sc_i,
                        int outer, int inner, float alpha, int relu)
{
    dim3 g((N + 31) / 32, (M + 31) / 32, outer * inner);
    wmma_gemm_kernel<<<g, 32, 0, st>>>(A, Bt, bias, C, M, N, K, lda, ldb, ldc,
                                       sa_o, sa_i, sb_o, sb_i, sc_o, sc_i,
                                       inner, alpha, relu);
}

extern "C" void kernel_launch(void* const* d_in, const int* in_sizes, int n_in,
                              void* d_out, int out_size, void* d_ws, size_t ws_size,
                              hipStream_t stream)
{
    (void)in_sizes; (void)n_in; (void)out_size; (void)ws_size;
    const int B = 32, S = 300, D = 512, L = 49, H = 8;
    const long long BS    = (long long)B * S;          // 9600
    const long long BSD   = BS * D;                    // 4,915,200
    const long long BS256 = BS * 256;
    const long long BHSS  = (long long)B * H * S * S;  // 23,040,000
    const int SCP = 52;                                // padded pitch for L=49

    const float* oq     = (const float*)d_in[0];
    const float* enc    = (const float*)d_in[1];
    const float* crd    = (const float*)d_in[2];
    const float* pos    = (const float*)d_in[3];
    const float* sinemb = (const float*)d_in[4];
    const float* W[43];
    for (int i = 5; i < 43; ++i) W[i] = (const float*)d_in[i];
    const float* b_sa_out = W[11];
    const float* b_pa_out = W[13];
    const float* ln1_g = W[14]; const float* ln1_b = W[15];
    const float* ln2_g = W[16]; const float* ln2_b = W[17];
    float* out = (float*)d_out;
    float* ws  = (float*)d_ws;

    // ---- workspace: o | transposed-weight arena | overlapping phase arena ----
    float* o = ws;
    float* wp = ws + BSD;
    float* wt_sa_q_obj = wp; wp += 512 * 512;
    float* wt_sa_q_pos = wp; wp += 512 * 256;
    float* wt_sa_k_obj = wp; wp += 512 * 512;
    float* wt_sa_k_pos = wp; wp += 512 * 256;
    float* wt_sa_v_obj = wp; wp += 512 * 512;
    float* wt_sa_out   = wp; wp += 512 * 512;
    float* wt_pa_out   = wp; wp += 512 * 512;
    float* wt_ca_q_obj = wp; wp += 512 * 512;
    float* wt_ca_q_pos = wp; wp += 512 * 256;
    float* wt_ca_k_enc = wp; wp += 512 * 256;
    float* wt_ca_k_pos = wp; wp += 512 * 256;
    float* wt_ca_v_enc = wp; wp += 512 * 512;
    float* wt_br[2][3];  // per branch: w_ao^T, w_d1^T, w_d2^T
    for (int br = 0; br < 2; ++br) {
        wt_br[br][0] = wp; wp += 512 * 256;   // (256,512)
        wt_br[br][1] = wp; wp += 256 * 1024;  // (1024,256)
        wt_br[br][2] = wp; wp += 1024 * 256;  // (256,1024)
    }
    float* P = wp;  // phase arena
    // phase 1
    float* q      = P;
    float* k      = q + BSD;
    float* v      = k + BSD;
    float* qp     = v + BSD;
    float* kp     = qp + BS256;
    float* scores = kp + BS256;      // (B,H,S,S); becomes probs1 in place
    float* probs2 = scores + BHSS;
    float* out1c  = probs2 + BHSS;
    float* out2c  = out1c + BSD;
    float* o1     = out2c + BSD;
    float* o2     = o1 + BSD;
    float* vt     = o2 + BSD;        // (B,H,64,S)
    // phase 2 (reuses arena)
    float* qobj   = P;
    float* qp2    = qobj + BSD;
    float* qcls   = qp2 + BS256;
    float* qreg   = qcls + BSD;
    float* encpos = qreg + BSD;                        // (49,512)
    float* kenc   = encpos + (long long)L * D;         // (B*49,256)
    float* kposp  = kenc + (long long)B * L * 256;     // (49,256)
    float* k2     = kposp + (long long)L * 256;        // (B,49,512)
    float* v2     = k2 + (long long)B * L * D;         // (B,49,512)
    float* scs    = v2 + (long long)B * L * D;         // (B,S,52)
    float* ca     = scs + BS * SCP;                    // (B,S,512)
    float* cab    = ca + BSD;                          // (B,S,256)
    float* xb     = cab + BS256;                       // (B,S,256)
    float* h1     = xb + BS256;                        // (B,S,1024)
    float* x2b    = h1 + BS * 1024;                    // (B,S,256)
    float* v2t    = x2b + BS256;                       // (B,512,52)

    const int TB = 256;
    auto T = [&](const float* Win, float* Wt, int Kd, int Nd) {
        long long tot = (long long)Kd * Nd;
        transpose_kernel<<<dim3((unsigned)((tot + TB - 1) / TB)), TB, 0, stream>>>(
            Win, Wt, Kd, Nd);
    };
    // ---- weight transposes (K,N) -> (N,K) ----
    T(W[5],  wt_sa_q_obj, 512, 512);  T(W[6],  wt_sa_q_pos, 512, 256);
    T(W[7],  wt_sa_k_obj, 512, 512);  T(W[8],  wt_sa_k_pos, 512, 256);
    T(W[9],  wt_sa_v_obj, 512, 512);  T(W[10], wt_sa_out,   512, 512);
    T(W[12], wt_pa_out,   512, 512);  T(W[18], wt_ca_q_obj, 512, 512);
    T(W[19], wt_ca_q_pos, 512, 256);  T(W[20], wt_ca_k_enc, 512, 256);
    T(W[21], wt_ca_k_pos, 512, 256);  T(W[22], wt_ca_v_enc, 512, 512);
    for (int br = 0; br < 2; ++br) {
        int pb = 23 + br * 10;
        T(W[pb + 0], wt_br[br][0], 512, 256);
        T(W[pb + 2], wt_br[br][1], 256, 1024);
        T(W[pb + 4], wt_br[br][2], 1024, 256);
    }

    // ================= self / pair attention =================
    gemm(stream, pos, wt_sa_q_pos, nullptr, qp, (int)BS, 256, 512, 512, 512, 256,
         0, 0, 0, 0, 0, 0, 1, 1, 1.f, 0);
    gemm(stream, pos, wt_sa_k_pos, nullptr, kp, (int)BS, 256, 512, 512, 512, 256,
         0, 0, 0, 0, 0, 0, 1, 1, 1.f, 0);
    gemm(stream, oq, wt_sa_q_obj, nullptr, q, (int)BS, 512, 512, 512, 512, 512,
         0, 0, 0, 0, 0, 0, 1, 1, 1.f, 0);
    gemm(stream, oq, wt_sa_k_obj, nullptr, k, (int)BS, 512, 512, 512, 512, 512,
         0, 0, 0, 0, 0, 0, 1, 1, 1.f, 0);
    gemm(stream, oq, wt_sa_v_obj, nullptr, v, (int)BS, 512, 512, 512, 512, 512,
         0, 0, 0, 0, 0, 0, 1, 1, 1.f, 0);
    add_dup_kernel<<<dim3((unsigned)((BSD + TB - 1) / TB)), TB, 0, stream>>>(q, qp, BSD);
    add_dup_kernel<<<dim3((unsigned)((BSD + TB - 1) / TB)), TB, 0, stream>>>(k, kp, BSD);

    // scores[b,h] = (q[b,h] @ k[b,h]^T) / 8 ; k is already N-major for this GEMM
    gemm(stream, q, k, nullptr, scores, S, S, 64, 512, 512, S,
         (long long)S * D, 64, (long long)S * D, 64,
         (long long)H * S * S, (long long)S * S, B, H, 0.125f, 0);

    softmax_pair_kernel<<<dim3(S, H, B), 128, 0, stream>>>(scores, probs2, crd, H, S);

    // v -> per-head K-major (64 x S) for probs @ v
    {
        long long tot = (long long)B * H * 64 * S;
        transpose_head_kernel<<<dim3((unsigned)((tot + TB - 1) / TB)), TB, 0, stream>>>(
            v, vt, B, H, S, D);
    }
    gemm(stream, scores, vt, nullptr, out1c, S, 64, S, S, S, 512,
         (long long)H * S * S, (long long)S * S, (long long)H * 64 * S, (long long)64 * S,
         (long long)S * D, 64, B, H, 1.f, 0);
    gemm(stream, probs2, vt, nullptr, out2c, S, 64, S, S, S, 512,
         (long long)H * S * S, (long long)S * S, (long long)H * 64 * S, (long long)64 * S,
         (long long)S * D, 64, B, H, 1.f, 0);

    gemm(stream, out1c, wt_sa_out, b_sa_out, o1, (int)BS, 512, 512, 512, 512, 512,
         0, 0, 0, 0, 0, 0, 1, 1, 1.f, 0);
    gemm(stream, out2c, wt_pa_out, b_pa_out, o2, (int)BS, 512, 512, 512, 512, 512,
         0, 0, 0, 0, 0, 0, 1, 1, 1.f, 0);

    fuse_ln_kernel<<<dim3((unsigned)BS), 128, 0, stream>>>(
        oq, o1, o2, ln1_g, ln1_b, ln2_g, ln2_b, o, D);

    // ================= cross attention =================
    sine_embed_kernel<<<dim3((unsigned)((L * (D / 2) + TB - 1) / TB)), TB, 0, stream>>>(
        encpos, L, D);

    gemm(stream, o, wt_ca_q_obj, nullptr, qobj, (int)BS, 512, 512, 512, 512, 512,
         0, 0, 0, 0, 0, 0, 1, 1, 1.f, 0);
    gemm(stream, sinemb, wt_ca_q_pos, nullptr, qp2, (int)BS, 256, 512, 512, 512, 256,
         0, 0, 0, 0, 0, 0, 1, 1, 1.f, 0);
    concat_heads_kernel<<<dim3((unsigned)((BSD + TB - 1) / TB)), TB, 0, stream>>>(
        qobj, 512, qp2, 0, qcls, BSD);
    concat_heads_kernel<<<dim3((unsigned)((BSD + TB - 1) / TB)), TB, 0, stream>>>(
        qobj + 256, 512, qp2, 0, qreg, BSD);

    gemm(stream, enc, wt_ca_k_enc, nullptr, kenc, B * L, 256, 512, 512, 512, 256,
         0, 0, 0, 0, 0, 0, 1, 1, 1.f, 0);
    gemm(stream, encpos, wt_ca_k_pos, nullptr, kposp, L, 256, 512, 512, 512, 256,
         0, 0, 0, 0, 0, 0, 1, 1, 1.f, 0);
    {
        long long tot = (long long)B * L * D;
        concat_heads_kernel<<<dim3((unsigned)((tot + TB - 1) / TB)), TB, 0, stream>>>(
            kenc, 256, kposp, L, k2, tot);
    }
    gemm(stream, enc, wt_ca_v_enc, nullptr, v2, B * L, 512, 512, 512, 512, 512,
         0, 0, 0, 0, 0, 0, 1, 1, 1.f, 0);
    {
        long long tot = (long long)B * D * SCP;
        transpose_b_kernel<<<dim3((unsigned)((tot + TB - 1) / TB)), TB, 0, stream>>>(
            v2, v2t, B, L, D, SCP);
    }

    // ================= cls / reg branches =================
    const float inv_sqrt_d = 0.04419417382415922f;  // 1/sqrt(512)
    for (int br = 0; br < 2; ++br) {
        int pb = 23 + br * 10;
        const float* b_ao = W[pb + 1];
        const float* b_d1 = W[pb + 3];
        const float* b_d2 = W[pb + 5];
        const float* g1   = W[pb + 6];
        const float* b1   = W[pb + 7];
        const float* g2   = W[pb + 8];
        const float* b2   = W[pb + 9];
        const float* qb = (br == 0) ? qcls : qreg;
        int off = br * 256;

        // scs[b] = (qb[b] @ k2[b]^T) / sqrt(512) ; k2 already N-major
        gemm(stream, qb, k2, nullptr, scs, S, L, 512, 512, 512, SCP,
             (long long)S * D, 0, (long long)L * D, 0,
             (long long)S * SCP, 0, B, 1, inv_sqrt_d, 0);
        softmax_row_kernel<<<dim3((unsigned)BS), 64, 0, stream>>>(scs, L, SCP);
        // ca[b] = scs[b] @ v2[b]  (v2t is K-major, pitch SCP)
        gemm(stream, scs, v2t, nullptr, ca, S, 512, L, SCP, SCP, 512,
             (long long)S * SCP, 0, (long long)D * SCP, 0,
             (long long)S * D, 0, B, 1, 1.f, 0);
        // out-proj to 256
        gemm(stream, ca, wt_br[br][0], b_ao, cab, (int)BS, 256, 512, 512, 512, 256,
             0, 0, 0, 0, 0, 0, 1, 1, 1.f, 0);
        // x = LN(o_half + cab)
        ln_res_kernel<<<dim3((unsigned)BS), 128, 0, stream>>>(
            o + off, D, cab, 256, g1, b1, xb, 256, 256);
        // FFN 256 -> 1024 (ReLU) -> 256
        gemm(stream, xb, wt_br[br][1], b_d1, h1, (int)BS, 1024, 256, 256, 256, 1024,
             0, 0, 0, 0, 0, 0, 1, 1, 1.f, 1);
        gemm(stream, h1, wt_br[br][2], b_d2, x2b, (int)BS, 256, 1024, 1024, 1024, 256,
             0, 0, 0, 0, 0, 0, 1, 1, 1.f, 0);
        // LN(x + x2) -> final output half
        ln_res_kernel<<<dim3((unsigned)BS), 128, 0, stream>>>(
            xb, 256, x2b, 256, g2, b2, out + off, D, 256);
    }
}
